// PPDocLayoutV2PositionRelationEmbedding_1760936592078
// MI455X (gfx1250) — matrix operand order repaired and checked by reading, not compiled, not run on hardware
//
#include <hip/hip_runtime.h>

// Reference: PPDocLayoutV2 position-relation embedding.
//   out[b,o,n,m] = sum_c pe[b,n,m,c] * W[o,c] + bias[o]
//   pe = [sin|cos](rel * 100 * 10^-f), rel = 4 box-relation comps, c = r*16 + sc*8 + f
// Store-bandwidth-bound (67MB out); contraction done with v_wmma_f32_16x16x32_f16.

typedef __attribute__((ext_vector_type(16))) _Float16 v16h;
typedef __attribute__((ext_vector_type(8)))  float    v8f;

#define BB     2
#define NN     1024
#define MM     1024
#define HEADS  8
#define FEAT   64          // 4 * EMBED_DIM
#define TPW    4           // 16-wide m-tiles per wave
#define EPSV   1e-5f

__global__ __launch_bounds__(256)
void posrel_wmma_kernel(const float4* __restrict__ src,   // (B,N,4)
                        const float4* __restrict__ tgt,   // (B,M,4)
                        const float*  __restrict__ W,     // (8,64)
                        const float*  __restrict__ bias,  // (8,)
                        float*        __restrict__ out)   // (B,8,N,M)
{
    const int lane = threadIdx.x & 31;
    const int wave = blockIdx.x * (blockDim.x >> 5) + (threadIdx.x >> 5);

    // wave -> (b, n, m-group of 64)
    const int mgrp = wave & 15;            // 16 groups of TPW*16 = 64 m each
    const int n    = (wave >> 4) & (NN - 1);
    const int b    = wave >> 14;

    const int p   = lane & 15;             // A row (head) / B column (pair-in-tile)
    const int grp = lane >> 4;             // K-half selector

    // ---- A fragments: W (8x64) padded to 16 rows, f16, two K=32 steps ----
    // A layout (16-bit 16x32): lane holds row (lane&15); within kstep,
    // feature K(h) = grp*16 + h.
    v16h a0 = {}; v16h a1 = {};
    if (p < HEADS) {
        const float* wr = W + p * FEAT;
        #pragma unroll
        for (int h = 0; h < 16; ++h) {
            const int c0 = (grp << 4) + h;
            a0[h] = (_Float16)wr[c0];
            a1[h] = (_Float16)wr[c0 + 32];
        }
    }

    // Bias-seeded accumulator template (valid half: lanes 0-15 / heads 0-7).
    v8f cinit;
    #pragma unroll
    for (int g = 0; g < 8; ++g) cinit[g] = bias[g];

    // Source box (uniform across wave); this lane only ever needs component
    // pair selected by grp: (x,w) for grp=0, (y,h) for grp=1.
    const float4 s  = src[b * NN + n];
    const float  cs = grp ? s.y : s.x;
    const float  ds = (grp ? s.w : s.z) + EPSV;

    // inv_freq[f] = 10000^(-f/4) = 10^-f
    const float invf[8] = {1.f, 1e-1f, 1e-2f, 1e-3f, 1e-4f, 1e-5f, 1e-6f, 1e-7f};

    float* outb = out + ((size_t)(b * HEADS) * NN + n) * MM;

    #pragma unroll
    for (int t = 0; t < TPW; ++t) {
        const int mb = (mgrp * TPW + t) * 16;
        const int m  = mb + p;

        const float4 tb = tgt[b * MM + m];
        const float  ct = grp ? tb.y : tb.x;
        const float  dt = (grp ? tb.w : tb.z) + EPSV;

        // rel[r]*SCALE for this lane's two components: r = grp (kstep0), 2+grp (kstep1)
        const float ra = __logf(__builtin_fabsf(cs - ct) / ds + 1.0f) * 100.0f;
        const float rb = __logf(ds / dt) * 100.0f;

        // B fragments: lane holds column p, K = grp*16 + h (per kstep);
        // block [16q,16q+16) = component r=q: sin(f=0..7) then cos(f=0..7).
        v16h b0, b1;
        #pragma unroll
        for (int f = 0; f < 8; ++f) {
            const float e0 = ra * invf[f];
            const float e1 = rb * invf[f];
            b0[f]     = (_Float16)__sinf(e0);
            b0[f + 8] = (_Float16)__cosf(e0);
            b1[f]     = (_Float16)__sinf(e1);
            b1[f + 8] = (_Float16)__cosf(e1);
        }

        v8f c = cinit;
        c = __builtin_amdgcn_wmma_f32_16x16x32_f16(false, a0, false, b0,
                                                   (short)0, c, false, false);
        c = __builtin_amdgcn_wmma_f32_16x16x32_f16(false, a1, false, b1,
                                                   (short)0, c, false, false);

        // C layout: VGPR g, lanes 0-15 -> out[head=g, m=mb+lane]
        if (lane < 16) {
            #pragma unroll
            for (int g = 0; g < 8; ++g)
                outb[(size_t)g * NN * MM + mb + lane] = c[g];
        }
    }
}

extern "C" void kernel_launch(void* const* d_in, const int* in_sizes, int n_in,
                              void* d_out, int out_size, void* d_ws, size_t ws_size,
                              hipStream_t stream) {
    const float4* src  = (const float4*)d_in[0];   // source_boxes (2,1024,4)
    const float4* tgt  = (const float4*)d_in[1];   // target_boxes (2,1024,4)
    const float*  W    = (const float*)d_in[2];    // (8,64)
    const float*  bias = (const float*)d_in[3];    // (8,)
    float*        out  = (float*)d_out;            // (2,8,1024,1024)

    // waves = B * N * (M / (16*TPW)) = 2*1024*16 = 32768 ; 8 waves/block
    const int blocks = (BB * NN * (MM / (16 * TPW)) * 32) / 256;   // 4096
    posrel_wmma_kernel<<<blocks, 256, 0, stream>>>(src, tgt, W, bias, out);
}
